// MultiHeadAttentionBlock_13563506720811
// MI455X (gfx1250) — matrix operand order repaired and checked by reading, not compiled, not run on hardware
//
#include <hip/hip_runtime.h>

// ---------------------------------------------------------------------------
// MHA block for MI455X (gfx1250): bf16 WMMA everywhere, f32 accumulate.
// GEMMs use async global->LDS staging (ASYNCcnt) when the toolchain has it.
// ---------------------------------------------------------------------------

typedef __bf16 bf16_t;
typedef __attribute__((ext_vector_type(16))) __bf16 v16bf;
typedef __attribute__((ext_vector_type(8)))  __bf16 v8bf;
typedef __attribute__((ext_vector_type(4)))  __bf16 v4bf;
typedef __attribute__((ext_vector_type(8)))  float  v8f;
typedef int v4i __attribute__((vector_size(16)));   // matches builtin param type

constexpr int Bn = 2, Sn = 2048, Dn = 1024, Hn = 16, DKn = 64;
constexpr int Mn  = Bn * Sn;   // 4096 rows for every GEMM in this block
constexpr int LDP = 72;        // flash LDS row stride (bf16) - bank-conflict pad
constexpr int LDK = 40;        // GEMM LDS K-stride (32 + 8 pad)

#if __has_builtin(__builtin_amdgcn_global_load_async_to_lds_b128)
#define HAVE_ASYNC_LDS 1
#else
#define HAVE_ASYNC_LDS 0
#endif

__device__ __forceinline__ void wait_async_0() {
#if __has_builtin(__builtin_amdgcn_s_wait_asynccnt)
  __builtin_amdgcn_s_wait_asynccnt(0);
#else
  asm volatile("s_wait_asynccnt 0" ::: "memory");
#endif
}

#if HAVE_ASYNC_LDS
// One 16-byte async global->LDS copy (tracked by ASYNCcnt, bypasses VGPRs).
__device__ __forceinline__ void async_cp16(const bf16_t* g, bf16_t* l) {
  bf16_t* gnc = const_cast<bf16_t*>(g);
  __builtin_amdgcn_global_load_async_to_lds_b128(
      (__attribute__((address_space(1))) v4i*)gnc,
      (__attribute__((address_space(3))) v4i*)l, 0, 0);
}
#endif

__device__ __forceinline__ v8f v8f_zero() {
  v8f z;
#pragma unroll
  for (int i = 0; i < 8; ++i) z[i] = 0.0f;
  return z;
}

__device__ __forceinline__ v8f wmma_bf16(v16bf a, v16bf b, v8f c) {
  // D = A(16x32 bf16) * B(32x16 bf16) + C(16x16 f32)
  return __builtin_amdgcn_wmma_f32_16x16x32_bf16(false, a, false, b,
                                                 (short)0, c, false, false);
}

// A-fragment (16x32, MxK), contiguous K order (matches load_bfrag):
//   lanes 0-15 : row M=lane, K = k0..k0+15
//   lanes 16-31: row M=lane-16, K = k0+16..k0+31
__device__ __forceinline__ v16bf load_afrag(const bf16_t* base, int ld) {
  const int lane = threadIdx.x & 31;
  const bf16_t* p = base + (size_t)(lane & 15) * ld + ((lane >> 4) << 4);
  v8bf lo = *(const v8bf*)(p);        // K = kb .. kb+7
  v8bf hi = *(const v8bf*)(p + 8);    // K = kb+8 .. kb+15
  return __builtin_shufflevector(lo, hi, 0, 1, 2, 3, 4, 5, 6, 7,
                                         8, 9, 10, 11, 12, 13, 14, 15);
}

// B-fragment (32x16, KxN): lane holds column n = lane&15; lanes 0-15 rows
// K=0..15, lanes 16-31 rows K=16..31. `base` = column n's storage row
// (row-major with 16 contiguous K values) -> one 32B load.
__device__ __forceinline__ v16bf load_bfrag(const bf16_t* base, int ld) {
  const int lane = threadIdx.x & 31;
  return *(const v16bf*)(base + (size_t)(lane & 15) * ld + ((lane >> 4) << 4));
}

// ---------------------------------------------------------------------------
// f32 -> bf16 elementwise (vector of 4)
// ---------------------------------------------------------------------------
__global__ void k_cvt_bf16(const float* __restrict__ in,
                           bf16_t* __restrict__ out, int n4) {
  int i = blockIdx.x * blockDim.x + threadIdx.x;
  if (i < n4) {
    float4 f = ((const float4*)in)[i];
    v4bf o;
    o[0] = (bf16_t)f.x; o[1] = (bf16_t)f.y;
    o[2] = (bf16_t)f.z; o[3] = (bf16_t)f.w;
    ((v4bf*)out)[i] = o;
  }
}

// ---------------------------------------------------------------------------
// C = A[M=4096,K=1024] * W^T + bias, W stored [N=1024, K=1024] row-major.
// Block = 4 waves, each wave a 64x64 tile (2x2 waves -> 128x128 per block).
// K-tiles (128x32 of A, 128x32 of W) stream through double-buffered LDS via
// GLOBAL_LOAD_ASYNC_TO_LDS_B128 when available; else direct global frags.
// OUT_MODE: 0 = bf16 row-major [M,D], 1 = bf16 transposed per head
//           Vt[b,h,d,s], 2 = f32 row-major (final output, + bias)
// ---------------------------------------------------------------------------
template <int OUT_MODE>
__global__ __launch_bounds__(128)
void gemm_xWT(const bf16_t* __restrict__ A, const bf16_t* __restrict__ W,
              const float* __restrict__ bias, void* __restrict__ out) {
  const int wid  = threadIdx.x >> 5;
  const int lane = threadIdx.x & 31;
  const int rowB = blockIdx.x * 128;
  const int colB = blockIdx.y * 128;
  const int wm = (wid >> 1) * 64, wn = (wid & 1) * 64;
  const int row0 = rowB + wm,  col0 = colB + wn;

  v8f acc[4][4];
#pragma unroll
  for (int mf = 0; mf < 4; ++mf)
#pragma unroll
    for (int nf = 0; nf < 4; ++nf) acc[mf][nf] = v8f_zero();

#if HAVE_ASYNC_LDS
  // -------- async LDS staging path (double buffered) --------
  __shared__ bf16_t sA[2][128 * LDK];
  __shared__ bf16_t sW[2][128 * LDK];
  const int t = threadIdx.x;  // thread t stages row t of each 128x32 tile

  auto stage = [&](const bf16_t* g, bf16_t* l) {
#pragma unroll
    for (int c = 0; c < 4; ++c)  // 4 x 16B = one 32-elem bf16 row
      async_cp16(g + c * 8, l + c * 8);
  };

  stage(A + (size_t)(rowB + t) * Dn, &sA[0][t * LDK]);
  stage(W + (size_t)(colB + t) * Dn, &sW[0][t * LDK]);
  wait_async_0();
  __syncthreads();

  for (int k0 = 0; k0 < Dn; k0 += 32) {
    const int cur = (k0 >> 5) & 1, nxt = cur ^ 1;
    if (k0 + 32 < Dn) {  // prefetch next K-tile while we compute
      stage(A + (size_t)(rowB + t) * Dn + k0 + 32, &sA[nxt][t * LDK]);
      stage(W + (size_t)(colB + t) * Dn + k0 + 32, &sW[nxt][t * LDK]);
    }

    v16bf af[4], bfr[4];
#pragma unroll
    for (int mf = 0; mf < 4; ++mf)
      af[mf] = load_afrag(&sA[cur][(wm + mf * 16) * LDK], LDK);
#pragma unroll
    for (int nf = 0; nf < 4; ++nf)
      bfr[nf] = load_bfrag(&sW[cur][(wn + nf * 16) * LDK], LDK);

#pragma unroll
    for (int mf = 0; mf < 4; ++mf)
#pragma unroll
      for (int nf = 0; nf < 4; ++nf)
        acc[mf][nf] = wmma_bf16(af[mf], bfr[nf], acc[mf][nf]);

    wait_async_0();   // next tile resident in LDS
    __syncthreads();  // all waves done reading cur / staging nxt
  }
#else
  // -------- fallback: direct global fragment loads (L2-resident weights) ----
  for (int k0 = 0; k0 < Dn; k0 += 32) {
    v16bf af[4], bfr[4];
#pragma unroll
    for (int mf = 0; mf < 4; ++mf)
      af[mf] = load_afrag(A + (size_t)(row0 + mf * 16) * Dn + k0, Dn);
#pragma unroll
    for (int nf = 0; nf < 4; ++nf)
      bfr[nf] = load_bfrag(W + (size_t)(col0 + nf * 16) * Dn + k0, Dn);

    if (k0 + 32 < Dn) {  // global_prefetch_b8 hint for next K tile
      __builtin_prefetch(A + (size_t)(row0 + lane) * Dn + k0 + 32, 0, 0);
      __builtin_prefetch(W + (size_t)(col0 + lane) * Dn + k0 + 32, 0, 0);
    }

#pragma unroll
    for (int mf = 0; mf < 4; ++mf)
#pragma unroll
      for (int nf = 0; nf < 4; ++nf)
        acc[mf][nf] = wmma_bf16(af[mf], bfr[nf], acc[mf][nf]);
  }
#endif

  const int hi = lane >> 4, nl = lane & 15;
#pragma unroll
  for (int mf = 0; mf < 4; ++mf)
#pragma unroll
    for (int nf = 0; nf < 4; ++nf) {
      const int n = col0 + nf * 16 + nl;
      const float bv = bias[n];
#pragma unroll
      for (int r = 0; r < 8; ++r) {
        const int m = row0 + mf * 16 + r + 8 * hi;
        const float v = acc[mf][nf][r] + bv;
        if constexpr (OUT_MODE == 0) {
          ((bf16_t*)out)[(size_t)m * Dn + n] = (bf16_t)v;
        } else if constexpr (OUT_MODE == 1) {
          const int b = m >> 11, s = m & (Sn - 1);
          const int h = n >> 6,  d = n & (DKn - 1);
          ((bf16_t*)out)[(size_t)((b * Hn + h) * DKn + d) * Sn + s] = (bf16_t)v;
        } else {
          ((float*)out)[(size_t)m * Dn + n] = v;
        }
      }
    }
}

// ---------------------------------------------------------------------------
// Flash attention: 1 wave per 64-query block of one (b,h). Online softmax.
// Q,K: [B,S,D] bf16 (head = 64-col slice); V pre-transposed Vt[b,h,d,s].
// ---------------------------------------------------------------------------
__global__ __launch_bounds__(128)
void flash_attn(const bf16_t* __restrict__ Q, const bf16_t* __restrict__ Kp,
                const bf16_t* __restrict__ Vt, bf16_t* __restrict__ O) {
  __shared__ bf16_t lds[4][64 * LDP];

  const int wid  = threadIdx.x >> 5;
  const int lane = threadIdx.x & 31;
  const int gw   = blockIdx.x * 4 + wid;
  const int qb   = gw & 31;       // 32 query blocks per (b,h)
  const int bh   = gw >> 5;
  const int b    = bh >> 4;
  const int h    = bh & 15;

  const bf16_t* Qh = Q  + (size_t)(b * Sn + qb * 64) * Dn + h * DKn;
  const bf16_t* Kh = Kp + (size_t)(b * Sn) * Dn + h * DKn;
  const bf16_t* Vh = Vt + (size_t)((b * Hn + h) * DKn) * Sn;
  bf16_t*       Oh = O  + (size_t)(b * Sn + qb * 64) * Dn + h * DKn;

  // Q fragments stay resident: 4 row-frags x 2 K-steps (DK=64)
  v16bf qf[4][2];
#pragma unroll
  for (int mf = 0; mf < 4; ++mf)
#pragma unroll
    for (int ks = 0; ks < 2; ++ks)
      qf[mf][ks] = load_afrag(Qh + (size_t)(mf * 16) * Dn + ks * 32, Dn);

  v8f o[4][4];
  float mi[4][8], li[4][8];
#pragma unroll
  for (int mf = 0; mf < 4; ++mf) {
#pragma unroll
    for (int r = 0; r < 8; ++r) { mi[mf][r] = -1e30f; li[mf][r] = 0.0f; }
#pragma unroll
    for (int nf = 0; nf < 4; ++nf) o[mf][nf] = v8f_zero();
  }

  bf16_t* lp = lds[wid];
  const int nl = lane & 15, hi = lane >> 4;

  for (int kb = 0; kb < Sn / 64; ++kb) {
    // ---- S = Q K^T over a 64-key block ----
    v8f sc[4][4];
#pragma unroll
    for (int mf = 0; mf < 4; ++mf)
#pragma unroll
      for (int nf = 0; nf < 4; ++nf) sc[mf][nf] = v8f_zero();

#pragma unroll
    for (int ks = 0; ks < 2; ++ks) {
      v16bf kf[4];
#pragma unroll
      for (int nf = 0; nf < 4; ++nf)
        kf[nf] = load_bfrag(Kh + (size_t)(kb * 64 + nf * 16) * Dn + ks * 32, Dn);
#pragma unroll
      for (int mf = 0; mf < 4; ++mf)
#pragma unroll
        for (int nf = 0; nf < 4; ++nf)
          sc[mf][nf] = wmma_bf16(qf[mf][ks], kf[nf], sc[mf][nf]);
    }

    // ---- scale + online softmax (rows live in the C-layout halves) ----
#pragma unroll
    for (int mf = 0; mf < 4; ++mf)
#pragma unroll
      for (int nf = 0; nf < 4; ++nf)
#pragma unroll
        for (int r = 0; r < 8; ++r) sc[mf][nf][r] *= 0.125f;  // 1/sqrt(64)

#pragma unroll
    for (int mf = 0; mf < 4; ++mf) {
      float nm[8], ps[8];
#pragma unroll
      for (int r = 0; r < 8; ++r)
        nm[r] = fmaxf(fmaxf(sc[mf][0][r], sc[mf][1][r]),
                      fmaxf(sc[mf][2][r], sc[mf][3][r]));
#pragma unroll
      for (int mask = 1; mask <= 8; mask <<= 1)
#pragma unroll
        for (int r = 0; r < 8; ++r)
          nm[r] = fmaxf(nm[r], __shfl_xor(nm[r], mask, 32));

#pragma unroll
      for (int r = 0; r < 8; ++r) {
        const float mnew = fmaxf(mi[mf][r], nm[r]);
        const float corr = __expf(mi[mf][r] - mnew);
        mi[mf][r] = mnew;
        li[mf][r] *= corr;
#pragma unroll
        for (int nf = 0; nf < 4; ++nf) o[mf][nf][r] *= corr;
      }

#pragma unroll
      for (int r = 0; r < 8; ++r) {
        float s = 0.0f;
#pragma unroll
        for (int nf = 0; nf < 4; ++nf) {
          const float p = __expf(sc[mf][nf][r] - mi[mf][r]);
          sc[mf][nf][r] = p;
          s += p;
        }
        ps[r] = s;
      }
#pragma unroll
      for (int mask = 1; mask <= 8; mask <<= 1)
#pragma unroll
        for (int r = 0; r < 8; ++r) ps[r] += __shfl_xor(ps[r], mask, 32);
#pragma unroll
      for (int r = 0; r < 8; ++r) li[mf][r] += ps[r];
    }

    // ---- P (C-layout) -> LDS -> A-layout for P*V; same wave, DS in-order ----
#pragma unroll
    for (int mf = 0; mf < 4; ++mf)
#pragma unroll
      for (int nf = 0; nf < 4; ++nf)
#pragma unroll
        for (int r = 0; r < 8; ++r)
          lp[(mf * 16 + r + 8 * hi) * LDP + nf * 16 + nl] =
              (bf16_t)sc[mf][nf][r];

#pragma unroll
    for (int ks = 0; ks < 2; ++ks) {
      v16bf vf[4], pf[4];
#pragma unroll
      for (int nf = 0; nf < 4; ++nf)  // Vt rows are d, contiguous in key idx
        vf[nf] = load_bfrag(Vh + (size_t)(nf * 16) * Sn + kb * 64 + ks * 32, Sn);
#pragma unroll
      for (int mf = 0; mf < 4; ++mf)
        pf[mf] = load_afrag(lp + (size_t)(mf * 16) * LDP + ks * 32, LDP);
#pragma unroll
      for (int mf = 0; mf < 4; ++mf)
#pragma unroll
        for (int nf = 0; nf < 4; ++nf)
          o[mf][nf] = wmma_bf16(pf[mf], vf[nf], o[mf][nf]);
    }
  }

  // ---- normalize + store bf16 [B,S,D] for the output projection ----
#pragma unroll
  for (int mf = 0; mf < 4; ++mf) {
    float inv[8];
#pragma unroll
    for (int r = 0; r < 8; ++r) inv[r] = 1.0f / li[mf][r];
#pragma unroll
    for (int nf = 0; nf < 4; ++nf)
#pragma unroll
      for (int r = 0; r < 8; ++r)
        Oh[(size_t)(mf * 16 + r + 8 * hi) * Dn + nf * 16 + nl] =
            (bf16_t)(o[mf][nf][r] * inv[r]);
  }
}

// ---------------------------------------------------------------------------
// Host orchestration
// ---------------------------------------------------------------------------
extern "C" void kernel_launch(void* const* d_in, const int* in_sizes, int n_in,
                              void* d_out, int out_size, void* d_ws,
                              size_t ws_size, hipStream_t stream) {
  (void)in_sizes; (void)n_in; (void)out_size; (void)ws_size;

  const float* q  = (const float*)d_in[0];
  const float* k  = (const float*)d_in[1];
  const float* v  = (const float*)d_in[2];
  const float* Wq = (const float*)d_in[3];
  const float* bq = (const float*)d_in[4];
  const float* Wk = (const float*)d_in[5];
  const float* bk = (const float*)d_in[6];
  const float* Wv = (const float*)d_in[7];
  const float* bv = (const float*)d_in[8];
  const float* Wo = (const float*)d_in[9];
  const float* bo = (const float*)d_in[10];
  // d_in[11] = mask: reference discards its masked_fill result -> unused.

  char* ws = (char*)d_ws;
  const size_t MB = 1024ull * 1024ull;
  bf16_t* xq  = (bf16_t*)(ws + 0 * MB);    // bf16 activations (8 MB each)
  bf16_t* xk  = (bf16_t*)(ws + 8 * MB);
  bf16_t* xv  = (bf16_t*)(ws + 16 * MB);
  bf16_t* wqb = (bf16_t*)(ws + 24 * MB);   // bf16 weights (2 MB each)
  bf16_t* wkb = (bf16_t*)(ws + 26 * MB);
  bf16_t* wvb = (bf16_t*)(ws + 28 * MB);
  bf16_t* wob = (bf16_t*)(ws + 30 * MB);
  bf16_t* Qp  = (bf16_t*)(ws + 32 * MB);   // projected Q [B,S,D]
  bf16_t* Kpj = (bf16_t*)(ws + 40 * MB);   // projected K [B,S,D]
  bf16_t* Vt  = (bf16_t*)(ws + 48 * MB);   // projected V transposed [B,H,DK,S]
  bf16_t* Ao  = (bf16_t*)(ws + 56 * MB);   // attention output [B,S,D]

  const int NA4 = (Mn * Dn) / 4;   // activation elements / 4
  const int NW4 = (Dn * Dn) / 4;   // weight elements / 4
  k_cvt_bf16<<<NA4 / 256, 256, 0, stream>>>(q,  xq,  NA4);
  k_cvt_bf16<<<NA4 / 256, 256, 0, stream>>>(k,  xk,  NA4);
  k_cvt_bf16<<<NA4 / 256, 256, 0, stream>>>(v,  xv,  NA4);
  k_cvt_bf16<<<NW4 / 256, 256, 0, stream>>>(Wq, wqb, NW4);
  k_cvt_bf16<<<NW4 / 256, 256, 0, stream>>>(Wk, wkb, NW4);
  k_cvt_bf16<<<NW4 / 256, 256, 0, stream>>>(Wv, wvb, NW4);
  k_cvt_bf16<<<NW4 / 256, 256, 0, stream>>>(Wo, wob, NW4);

  dim3 gg(Mn / 128, Dn / 128);  // (32, 8)
  gemm_xWT<0><<<gg, 128, 0, stream>>>(xq, wqb, bq, (void*)Qp);
  gemm_xWT<0><<<gg, 128, 0, stream>>>(xk, wkb, bk, (void*)Kpj);
  gemm_xWT<1><<<gg, 128, 0, stream>>>(xv, wvb, bv, (void*)Vt);

  flash_attn<<<(Bn * Hn * (Sn / 64)) / 4, 128, 0, stream>>>(Qp, Kpj, Vt, Ao);

  gemm_xWT<2><<<gg, 128, 0, stream>>>(Ao, wob, bo, d_out);
}
